// MetalSite_40011915329895
// MI455X (gfx1250) — matrix operand order repaired and checked by reading, not compiled, not run on hardware
//
#include <hip/hip_runtime.h>

// ---------------- problem constants ----------------
#define BB      4
#define NN      2000
#define CNODE   1024
#define HH      128
#define KNB     30
#define KP      32          // padded neighbor count for WMMA M-tiles
#define FFD     512
#define NLAYERS 4
#define NHEADS  4
#define DHEAD   32
#define LN_EPS  1e-5f
#define FMAXV   3.4028235e38f

typedef unsigned short ushort_t;
typedef __attribute__((ext_vector_type(16))) __bf16    v16bf;
typedef __attribute__((ext_vector_type(16))) ushort_t  v16us;
typedef __attribute__((ext_vector_type(8)))  ushort_t  v8us;
typedef __attribute__((ext_vector_type(8)))  float     v8f;

// f32 -> bf16 (round to nearest even)
__device__ __forceinline__ ushort_t f2b(float x) {
  unsigned u = __builtin_bit_cast(unsigned, x);
  return (ushort_t)((u + 0x7fffu + ((u >> 16) & 1u)) >> 16);
}

// ---- CDNA5 async global->LDS copy (16B per lane, tracked by ASYNCcnt) ----
// GLOBAL_LOAD_ASYNC_TO_LDS_B128: copies 16B from the global address in vaddr to the LDS byte address in vdst
__device__ __forceinline__ void async_g2l_b128(unsigned lds_addr, const void* gaddr) {
  asm volatile("global_load_async_to_lds_b128 %0, %1, off"
               :: "v"(lds_addr), "v"(gaddr) : "memory");
}
__device__ __forceinline__ void wait_async() {
  asm volatile("s_wait_asynccnt 0" ::: "memory");
}
// generic pointer to __shared__ -> raw LDS byte offset (flat aperture: addr[31:0])
__device__ __forceinline__ unsigned lds_off(const void* p) {
  return (unsigned)(size_t)p;
}

// A-tile loader (16x32 bf16, row-major src with stride ld elements).
// lane L (m=L&15, half=L>>4): elems 0..15 <- k=16*half+j (matches ldB)
__device__ __forceinline__ v16us ldA(const ushort_t* base, int ld) {
  int lane = threadIdx.x & 31;
  int m = lane & 15, half = lane >> 4;
  const ushort_t* p = base + m * ld + 16 * half;
  v8us a = *(const v8us*)p;
  v8us b = *(const v8us*)(p + 8);
  v16us r;
#pragma unroll
  for (int j = 0; j < 8; ++j) { r[j] = a[j]; r[j + 8] = b[j]; }
  return r;
}

// B-tile loader from row-major bf16 weight W[Nrows][K] (y = x @ W^T so B[k][n] = W[n][k]).
// Contiguous k order (same as ldA): B lanes 0-15 hold k=0..15, lanes 16-31 hold k=16..31, n = lane&15 -> one 32B load.
__device__ __forceinline__ v16us ldB(const ushort_t* W, int K, int ntile, int ktile) {
  int lane = threadIdx.x & 31;
  int n = lane & 15, half = lane >> 4;
  return *(const v16us*)(W + (size_t)(ntile * 16 + n) * K + ktile * 32 + half * 16);
}

__device__ __forceinline__ v8f wmma_bf(v16us a, v16us b, v8f c) {
  return __builtin_amdgcn_wmma_f32_16x16x32_bf16(
      false, __builtin_bit_cast(v16bf, a),
      false, __builtin_bit_cast(v16bf, b),
      (short)0, c, false, false);
}

// C store (f32 16x16): elem r -> row = 8*(lane>>4)+r, col = lane&15
__device__ __forceinline__ void stC(float* s, int ld, v8f a) {
  int lane = threadIdx.x & 31, n = lane & 15, mo = (lane >> 4) * 8;
#pragma unroll
  for (int r = 0; r < 8; ++r) s[(mo + r) * ld + n] = a[r];
}

__device__ __forceinline__ float wred(float v) {
#pragma unroll
  for (int o = 16; o; o >>= 1) v += __shfl_xor(v, o, 32);
  return v;
}

// ---------------- weight f32 -> bf16 ----------------
__global__ void k_cvt(const float* __restrict__ s, ushort_t* __restrict__ d, int n) {
  int i = blockIdx.x * blockDim.x + threadIdx.x;
  if (i < n) d[i] = f2b(s[i]);
}

// ---------------- top-k neighbor selection ----------------
__global__ void k_topk(const float* __restrict__ X, const float* __restrict__ mask,
                       int* __restrict__ Eidx, float* __restrict__ Dnb) {
  const int node = blockIdx.x;
  const int b = node / NN;
  const int t = threadIdx.x;
  __shared__ float sD[2048];
  __shared__ float rv[256];
  __shared__ int ri[256];
  const float xi0 = X[node * 3 + 0], xi1 = X[node * 3 + 1], xi2 = X[node * 3 + 2];
  const float mi = mask[node];
  float lmax = 0.f;
  for (int j = t; j < 2048; j += 256) {
    float d = FMAXV;
    if (j < NN) {
      const float* xj = X + (size_t)(b * NN + j) * 3;
      float dx = xi0 - xj[0], dy = xi1 - xj[1], dz = xi2 - xj[2];
      d = mi * mask[b * NN + j] * sqrtf(dx * dx + dy * dy + dz * dz + 1e-6f);
      lmax = fmaxf(lmax, d);
    }
    sD[j] = d;
  }
  rv[t] = lmax; __syncthreads();
  for (int s = 128; s; s >>= 1) { if (t < s) rv[t] = fmaxf(rv[t], rv[t + s]); __syncthreads(); }
  float dmax = rv[0]; __syncthreads();
  for (int j = t; j < NN; j += 256) {
    float m2 = mi * mask[b * NN + j];
    sD[j] = sD[j] + (1.f - m2) * dmax;     // D_adj
  }
  __syncthreads();
  for (int sel = 0; sel < KNB; ++sel) {
    float bv = FMAXV; int bi = 0x7fffffff;
    for (int j = t; j < 2048; j += 256) {
      float v = sD[j];
      if (v < bv || (v == bv && j < bi)) { bv = v; bi = j; }
    }
    rv[t] = bv; ri[t] = bi; __syncthreads();
    for (int s = 128; s; s >>= 1) {
      if (t < s && (rv[t + s] < rv[t] || (rv[t + s] == rv[t] && ri[t + s] < ri[t]))) {
        rv[t] = rv[t + s]; ri[t] = ri[t + s];
      }
      __syncthreads();
    }
    if (t == 0) {
      int j = ri[0];
      Eidx[node * KNB + sel] = j;
      Dnb[node * KNB + sel] = rv[0];
      sD[j] = FMAXV;
    }
    __syncthreads();
  }
}

// ---------------- edge embed: feats(32) -> W_edge -> LN -> W_e + b_e -> hE (bf16) ----------------
__global__ void k_edge(const int* __restrict__ Eidx, const float* __restrict__ Dnb,
                       const ushort_t* __restrict__ Wedgeb,
                       const float* __restrict__ lng, const float* __restrict__ lnb,
                       const ushort_t* __restrict__ Web, const float* __restrict__ b_e,
                       ushort_t* __restrict__ hE) {
  const int node = blockIdx.x;
  const int i = node % NN;
  const int t = threadIdx.x, lane = t & 31, w = t >> 5;
  __shared__ __align__(16) ushort_t sF[KP * 32];
  __shared__ float sY[KP * HH];
  __shared__ __align__(16) ushort_t sYb[KP * HH];
  for (int idx = t; idx < KP * 32; idx += 256) {
    int r = idx >> 5, c = idx & 31;
    float f = 0.f;
    if (r < KNB) {
      int j = Eidx[node * KNB + r];
      float dpos = (float)(j - i);
      if (c < 16) {
        int cc = (c < 8) ? c : c - 8;
        float freq = expf(-(float)cc * (9.210340371976184f / 8.f));  // 10000^(-cc/8)
        float ang = dpos * freq;
        f = (c < 8) ? cosf(ang) : sinf(ang);
      } else {
        float mu = 2.f + (20.f / 15.f) * (float)(c - 16);            // linspace(2,22,16)
        float u = (Dnb[node * KNB + r] - mu) * (1.f / 1.25f);        // sigma = 20/16
        f = expf(-u * u);
      }
    }
    sF[idx] = f2b(f);
  }
  __syncthreads();
  for (int mt = 0; mt < 2; ++mt) {                  // GEMM1: (32x32) @ (32->128), K=32 one step
    v8f acc = {};
    acc = wmma_bf(ldA(sF + mt * 16 * 32, 32), ldB(Wedgeb, 32, w, 0), acc);
    stC(sY + mt * 16 * HH + w * 16, HH, acc);
  }
  __syncthreads();
  for (int rr = 0; rr < 4; ++rr) {                  // LayerNorm per row
    int row = w + rr * 8;
    float xs[4], s = 0.f;
#pragma unroll
    for (int q = 0; q < 4; ++q) { int c = lane + 32 * q; xs[q] = sY[row * HH + c]; s += xs[q]; }
    s = wred(s);
    float mean = s * (1.f / HH), s2 = 0.f;
#pragma unroll
    for (int q = 0; q < 4; ++q) { float d = xs[q] - mean; s2 += d * d; }
    s2 = wred(s2);
    float inv = rsqrtf(s2 * (1.f / HH) + LN_EPS);
#pragma unroll
    for (int q = 0; q < 4; ++q) {
      int c = lane + 32 * q;
      sYb[row * HH + c] = f2b((xs[q] - mean) * inv * lng[c] + lnb[c]);
    }
  }
  __syncthreads();
  for (int mt = 0; mt < 2; ++mt) {                  // GEMM2: (32x128) @ (128->128) + b_e
    v8f acc = {};
#pragma unroll
    for (int kt = 0; kt < 4; ++kt)
      acc = wmma_bf(ldA(sYb + mt * 16 * HH + kt * 32, HH), ldB(Web, HH, w, kt), acc);
    int n = lane & 15, mo = (lane >> 4) * 8;
    int col = w * 16 + n;
    float bv = b_e[col];
#pragma unroll
    for (int r = 0; r < 8; ++r) {
      int row = mt * 16 + mo + r;
      float v = (row < KNB) ? acc[r] + bv : 0.f;    // zero pad rows
      hE[((size_t)node * KP + row) * HH + col] = f2b(v);
    }
  }
}

// ---------------- h_V = V @ W_v^T + b_v ----------------
__global__ void k_hv0(const float* __restrict__ V, const ushort_t* __restrict__ Wvb,
                      const float* __restrict__ b_v,
                      float* __restrict__ hV, ushort_t* __restrict__ hVb16) {
  const int g0 = blockIdx.x * 16;
  const int t = threadIdx.x, lane = t & 31, w = t >> 5;
  __shared__ __align__(16) ushort_t sV[16 * CNODE];
  for (int idx = t; idx < 16 * CNODE; idx += 256) {
    int r = idx >> 10, c = idx & 1023;
    sV[idx] = f2b(V[(size_t)(g0 + r) * CNODE + c]);
  }
  __syncthreads();
  v8f acc = {};
#pragma unroll 4
  for (int kt = 0; kt < 32; ++kt)
    acc = wmma_bf(ldA(sV + kt * 32, CNODE), ldB(Wvb, CNODE, w, kt), acc);
  int n = lane & 15, mo = (lane >> 4) * 8;
  int col = w * 16 + n;
  float bv = b_v[col];
#pragma unroll
  for (int r = 0; r < 8; ++r) {
    int gn = g0 + mo + r;
    float v = acc[r] + bv;
    hV[(size_t)gn * HH + col] = v;
    hVb16[(size_t)gn * HH + col] = f2b(v);
  }
}

// ---------------- fused attention layer (one node per block) ----------------
__global__ void k_attn(const float* __restrict__ hVin, const ushort_t* __restrict__ hVb16,
                       const ushort_t* __restrict__ hE, const int* __restrict__ Eidx,
                       const float* __restrict__ mask,
                       const ushort_t* __restrict__ WKb, const ushort_t* __restrict__ WVb,
                       const float* __restrict__ WQ, const float* __restrict__ WO,
                       const float* __restrict__ ln1g, const float* __restrict__ ln1b,
                       float* __restrict__ hVmid, ushort_t* __restrict__ hVmid16) {
  const int node = blockIdx.x;
  const int b = node / NN;
  const int t = threadIdx.x, w = t >> 5;
  __shared__ __align__(16) ushort_t sEV[KP * 256];   // h_EV tile (32x256 bf16)
  __shared__ float sKt[KP * HH];
  __shared__ float sVt[KP * HH];
  __shared__ float shv[HH], sq[HH], slog[NHEADS * 32], sattn[NHEADS * 32];
  __shared__ float souto[HH], sdh[HH], sred[HH];
  __shared__ float smsk[KP];
  __shared__ int snb[KP];
  if (t < KP) {
    int j = (t < KNB) ? Eidx[node * KNB + t] : 0;
    snb[t] = j;
    smsk[t] = (t < KNB) ? mask[b * NN + j] : 0.f;
  }
  if (t < HH) shv[t] = hVin[(size_t)node * HH + t];
  __syncthreads();
  // h_E half of h_EV via CDNA5 async global->LDS (512 x 16B chunks, 2 per thread)
  for (int q = t; q < (KP * HH) / 8; q += 256) {
    int r = q >> 4, c = (q & 15) * 8;
    async_g2l_b128(lds_off(&sEV[r * 256 + c]),
                   hE + ((size_t)node * KP + r) * HH + c);
  }
  // gathered-h_V half (irregular gather, regular stores)
  for (int idx = t; idx < KP * HH; idx += 256) {
    int r = idx >> 7, c = idx & 127;
    ushort_t v = 0;
    if (r < KNB) v = hVb16[((size_t)(b * NN + snb[r])) * HH + c];
    sEV[r * 256 + HH + c] = v;
  }
  wait_async();
  __syncthreads();
  for (int mt = 0; mt < 2; ++mt) {                   // Kt,Vt: (32x256) @ (256->128), share A
    v8f aK = {}, aV = {};
#pragma unroll
    for (int kt = 0; kt < 8; ++kt) {
      v16us a = ldA(sEV + mt * 16 * 256 + kt * 32, 256);
      aK = wmma_bf(a, ldB(WKb, 256, w, kt), aK);
      aV = wmma_bf(a, ldB(WVb, 256, w, kt), aV);
    }
    stC(sKt + mt * 16 * HH + w * 16, HH, aK);
    stC(sVt + mt * 16 * HH + w * 16, HH, aV);
  }
  __syncthreads();
  if (t < HH) {                                      // Q = h_V @ W_Q^T (small, VALU)
    float s = 0.f;
#pragma unroll 4
    for (int c = 0; c < HH; ++c) s += shv[c] * WQ[(size_t)t * HH + c];
    sq[t] = s * 0.17677669529663687f;                // 1/sqrt(32)
  }
  __syncthreads();
  float maski = mask[node];
  if (t < NHEADS * KNB) {                            // logits
    int h = t / KNB, k = t % KNB;
    float s = 0.f;
#pragma unroll 8
    for (int d = 0; d < DHEAD; ++d) s += sq[h * DHEAD + d] * sKt[k * HH + h * DHEAD + d];
    float m = maski * smsk[k];
    slog[h * 32 + k] = (m > 0.f) ? s : -FMAXV;
  }
  __syncthreads();
  if (t < NHEADS) {                                  // softmax per head
    float mx = -FMAXV;
    for (int k = 0; k < KNB; ++k) mx = fmaxf(mx, slog[t * 32 + k]);
    float sum = 0.f;
    for (int k = 0; k < KNB; ++k) sum += expf(slog[t * 32 + k] - mx);
    float invs = 1.f / sum;
    for (int k = 0; k < KNB; ++k)
      sattn[t * 32 + k] = expf(slog[t * 32 + k] - mx) * invs * (maski * smsk[k]);
  }
  __syncthreads();
  if (t < HH) {                                      // out = attn @ Vt
    int h = t >> 5;
    float s = 0.f;
#pragma unroll 6
    for (int k = 0; k < KNB; ++k) s += sattn[h * 32 + k] * sVt[k * HH + t];
    souto[t] = s;
  }
  __syncthreads();
  if (t < HH) {                                      // W_O (VALU)
    float s = 0.f;
#pragma unroll 4
    for (int c = 0; c < HH; ++c) s += souto[c] * WO[(size_t)t * HH + c];
    sdh[t] = s;
  }
  __syncthreads();
  float x = 0.f;                                     // residual + LN1
  if (t < HH) { x = shv[t] + sdh[t]; sred[t] = x; }
  __syncthreads();
  for (int s = 64; s; s >>= 1) { if (t < s) sred[t] += sred[t + s]; __syncthreads(); }
  float mean = sred[0] * (1.f / HH); __syncthreads();
  if (t < HH) { float d = x - mean; sred[t] = d * d; }
  __syncthreads();
  for (int s = 64; s; s >>= 1) { if (t < s) sred[t] += sred[t + s]; __syncthreads(); }
  float var = sred[0] * (1.f / HH);
  if (t < HH) {
    float y = (x - mean) * rsqrtf(var + LN_EPS) * ln1g[t] + ln1b[t];
    hVmid[(size_t)node * HH + t] = y;
    hVmid16[(size_t)node * HH + t] = f2b(y);
  }
}

// ---------------- FFN + residual + LN2 + mask (16 nodes / block) ----------------
__global__ void k_ffn(const float* __restrict__ hVmid, const ushort_t* __restrict__ hVmid16,
                      const ushort_t* __restrict__ Wfib, const float* __restrict__ bfi,
                      const ushort_t* __restrict__ Wfob, const float* __restrict__ bfo,
                      const float* __restrict__ ln2g, const float* __restrict__ ln2b,
                      const float* __restrict__ mask,
                      float* __restrict__ hVout, ushort_t* __restrict__ hVb16) {
  const int g0 = blockIdx.x * 16;
  const int t = threadIdx.x, lane = t & 31, w = t >> 5;
  __shared__ __align__(16) ushort_t sA[16 * HH];
  __shared__ __align__(16) ushort_t sMid[16 * FFD];
  __shared__ float sOut[16 * HH];
  // A-tile stage: one async 16B chunk per thread (16*128 bf16 = 256 chunks)
  async_g2l_b128(lds_off(&sA[t * 8]), hVmid16 + (size_t)g0 * HH + t * 8);
  wait_async();
  __syncthreads();
  int n = lane & 15, mo = (lane >> 4) * 8;
  for (int nt4 = 0; nt4 < 4; ++nt4) {                // GEMM1 (128->512) + bias + relu -> bf16
    int nt = w * 4 + nt4;
    v8f acc = {};
#pragma unroll
    for (int kt = 0; kt < 4; ++kt)
      acc = wmma_bf(ldA(sA + kt * 32, HH), ldB(Wfib, HH, nt, kt), acc);
    float bv = bfi[nt * 16 + n];
#pragma unroll
    for (int r = 0; r < 8; ++r) {
      float v = acc[r] + bv; v = v > 0.f ? v : 0.f;
      sMid[(mo + r) * FFD + nt * 16 + n] = f2b(v);
    }
  }
  __syncthreads();
  {                                                  // GEMM2 (512->128) + bias
    v8f acc = {};
#pragma unroll
    for (int kt = 0; kt < 16; ++kt)
      acc = wmma_bf(ldA(sMid + kt * 32, FFD), ldB(Wfob, FFD, w, kt), acc);
    float bv = bfo[w * 16 + n];
#pragma unroll
    for (int r = 0; r < 8; ++r) sOut[(mo + r) * HH + w * 16 + n] = acc[r] + bv;
  }
  __syncthreads();
  for (int rr = 0; rr < 2; ++rr) {                   // residual + LN2 + mask
    int row = w + rr * 8, gn = g0 + row;
    float xs[4], s = 0.f;
#pragma unroll
    for (int q = 0; q < 4; ++q) {
      int c = lane + 32 * q;
      xs[q] = hVmid[(size_t)gn * HH + c] + sOut[row * HH + c];
      s += xs[q];
    }
    s = wred(s);
    float mean = s * (1.f / HH), s2 = 0.f;
#pragma unroll
    for (int q = 0; q < 4; ++q) { float d = xs[q] - mean; s2 += d * d; }
    s2 = wred(s2);
    float inv = rsqrtf(s2 * (1.f / HH) + LN_EPS);
    float mk = mask[gn];
#pragma unroll
    for (int q = 0; q < 4; ++q) {
      int c = lane + 32 * q;
      float y = ((xs[q] - mean) * inv * ln2g[c] + ln2b[c]) * mk;
      hVout[(size_t)gn * HH + c] = y;
      hVb16[(size_t)gn * HH + c] = f2b(y);
    }
  }
}

// ---------------- output heads (ZN, CA, MG, MN order) ----------------
struct HeadArgs {
  const ushort_t* W1[4];
  const float* W2[4];
  const float* b1[4];
  const float* b2[4];
};

__global__ void k_heads(const ushort_t* __restrict__ hVb16, HeadArgs ha, float* __restrict__ out) {
  const int g0 = blockIdx.x * 16;
  const int t = threadIdx.x, lane = t & 31, w = t >> 5;
  __shared__ __align__(16) ushort_t sA[16 * HH];
  __shared__ float sE[16 * HH];
  async_g2l_b128(lds_off(&sA[t * 8]), hVb16 + (size_t)g0 * HH + t * 8);
  wait_async();
  __syncthreads();
  for (int m = 0; m < 4; ++m) {
    v8f acc = {};
#pragma unroll
    for (int kt = 0; kt < 4; ++kt)
      acc = wmma_bf(ldA(sA + kt * 32, HH), ldB(ha.W1[m], HH, w, kt), acc);
    int n = lane & 15, mo = (lane >> 4) * 8;
    float bv = ha.b1[m][w * 16 + n];
#pragma unroll
    for (int r = 0; r < 8; ++r) {
      float v = acc[r] + bv;
      v = v > 0.f ? v : expf(v) - 1.f;               // elu
      sE[(mo + r) * HH + w * 16 + n] = v;
    }
    __syncthreads();
    for (int rr = 0; rr < 2; ++rr) {                 // dot with W2 row + b2
      int row = w + rr * 8, gn = g0 + row;
      float s = 0.f;
#pragma unroll
      for (int q = 0; q < 4; ++q) { int c = lane + 32 * q; s += sE[row * HH + c] * ha.W2[m][c]; }
      s = wred(s);
      if (lane == 0) {
        int bb = gn / NN, ii = gn % NN;
        out[(size_t)bb * (4 * NN) + m * NN + ii] = s + ha.b2[m][0];
      }
    }
    __syncthreads();
  }
}

// ---------------- host ----------------
// Assumed flatten order (jax tree: dict keys sorted): V, X, mask, then params sorted:
// W_CA1,W_CA2,W_MG1,W_MG2,W_MN1,W_MN2,W_ZN1,W_ZN2,W_e,W_edge,W_v, b_CA1..b_ZN2,b_e,b_v,
// layers[0..3]{W_K,W_O,W_Q,W_V,W_fi,W_fo,b_fi,b_fo,ln1_b,ln1_g,ln2_b,ln2_g}, ln_e_b, ln_e_g
extern "C" void kernel_launch(void* const* d_in, const int* in_sizes, int n_in,
                              void* d_out, int out_size, void* d_ws, size_t ws_size,
                              hipStream_t stream) {
  (void)in_sizes; (void)n_in; (void)out_size; (void)ws_size;
  const float* V    = (const float*)d_in[0];
  const float* X    = (const float*)d_in[1];
  const float* mask = (const float*)d_in[2];

  char* ws = (char*)d_ws;
  size_t off = 0;
  auto alloc = [&](size_t bytes) -> char* {
    char* p = ws + off;
    off = (off + bytes + 255) & ~(size_t)255;
    return p;
  };
  const size_t NTOT = (size_t)BB * NN;
  int*      Eidx   = (int*)     alloc(NTOT * KNB * 4);
  float*    Dnb    = (float*)   alloc(NTOT * KNB * 4);
  ushort_t* hE     = (ushort_t*)alloc(NTOT * KP * HH * 2);
  float*    hVA    = (float*)   alloc(NTOT * HH * 4);
  float*    hVB    = (float*)   alloc(NTOT * HH * 4);
  ushort_t* hVb    = (ushort_t*)alloc(NTOT * HH * 2);
  float*    hVm    = (float*)   alloc(NTOT * HH * 4);
  ushort_t* hVm16  = (ushort_t*)alloc(NTOT * HH * 2);
  ushort_t* Wvb    = (ushort_t*)alloc((size_t)HH * CNODE * 2);
  ushort_t* Web    = (ushort_t*)alloc((size_t)HH * HH * 2);
  ushort_t* Wedgeb = (ushort_t*)alloc((size_t)HH * 32 * 2);
  ushort_t *WKb[NLAYERS], *WVb[NLAYERS], *Wfib[NLAYERS], *Wfob[NLAYERS], *W1b[4];
  for (int L = 0; L < NLAYERS; ++L) {
    WKb[L]  = (ushort_t*)alloc((size_t)HH * 256 * 2);
    WVb[L]  = (ushort_t*)alloc((size_t)HH * 256 * 2);
    Wfib[L] = (ushort_t*)alloc((size_t)FFD * HH * 2);
    Wfob[L] = (ushort_t*)alloc((size_t)HH * FFD * 2);
  }
  for (int m = 0; m < 4; ++m) W1b[m] = (ushort_t*)alloc((size_t)HH * HH * 2);

  auto cvt = [&](int idx, ushort_t* d, int nelem) {
    k_cvt<<<(nelem + 255) / 256, 256, 0, stream>>>((const float*)d_in[idx], d, nelem);
  };
  cvt(13, Wvb, HH * CNODE);    // W_v
  cvt(11, Web, HH * HH);       // W_e
  cvt(12, Wedgeb, HH * 32);    // W_edge
  for (int L = 0; L < NLAYERS; ++L) {
    int base = 24 + 12 * L;
    cvt(base + 0, WKb[L],  HH * 256);  // W_K
    cvt(base + 3, WVb[L],  HH * 256);  // W_V
    cvt(base + 4, Wfib[L], FFD * HH);  // W_fi
    cvt(base + 5, Wfob[L], HH * FFD);  // W_fo
  }
  const int i1[4] = {9, 3, 5, 7};      // W_ZN1, W_CA1, W_MG1, W_MN1 (forward order)
  const int i2[4] = {10, 4, 6, 8};
  const int j1[4] = {20, 14, 16, 18};
  const int j2[4] = {21, 15, 17, 19};
  for (int m = 0; m < 4; ++m) cvt(i1[m], W1b[m], HH * HH);

  k_topk<<<BB * NN, 256, 0, stream>>>(X, mask, Eidx, Dnb);
  k_edge<<<BB * NN, 256, 0, stream>>>(Eidx, Dnb, Wedgeb,
                                      (const float*)d_in[73], (const float*)d_in[72],
                                      Web, (const float*)d_in[22], hE);
  k_hv0<<<BB * NN / 16, 256, 0, stream>>>(V, Wvb, (const float*)d_in[23], hVA, hVb);

  float* cur = hVA; float* nxt = hVB;
  for (int L = 0; L < NLAYERS; ++L) {
    int base = 24 + 12 * L;
    k_attn<<<BB * NN, 256, 0, stream>>>(cur, hVb, hE, Eidx, mask, WKb[L], WVb[L],
        (const float*)d_in[base + 2],   // W_Q
        (const float*)d_in[base + 1],   // W_O
        (const float*)d_in[base + 9],   // ln1_g
        (const float*)d_in[base + 8],   // ln1_b
        hVm, hVm16);
    k_ffn<<<BB * NN / 16, 256, 0, stream>>>(hVm, hVm16, Wfib[L],
        (const float*)d_in[base + 6],   // b_fi
        Wfob[L],
        (const float*)d_in[base + 7],   // b_fo
        (const float*)d_in[base + 11],  // ln2_g
        (const float*)d_in[base + 10],  // ln2_b
        mask, nxt, hVb);
    float* tmp = cur; cur = nxt; nxt = tmp;
  }

  HeadArgs ha;
  for (int m = 0; m < 4; ++m) {
    ha.W1[m] = W1b[m];
    ha.W2[m] = (const float*)d_in[i2[m]];
    ha.b1[m] = (const float*)d_in[j1[m]];
    ha.b2[m] = (const float*)d_in[j2[m]];
  }
  k_heads<<<BB * NN / 16, 256, 0, stream>>>(hVb, ha, (float*)d_out);
}
